// BERT_LSTM_CRF_42760694399510
// MI455X (gfx1250) — matrix operand order repaired and checked by reading, not compiled, not run on hardware
//
#include <hip/hip_runtime.h>

// ---------------------------------------------------------------------------
// BERT_LSTM_CRF head for MI455X (gfx1250, wave32, WMMA + TDM + async-LDS).
// B=64 S=512 D=768 H=384 T=22.
//   K_gemm_xg: xg = embeds @ Wih^T + (bih+bhh)   [v_wmma_f32_16x16x32_f16]
//   K_scan:    512-step BiLSTM recurrence, 48 persistent WGs; Whh slice
//              loaded via TENSOR_LOAD_TO_LDS (TDM, padded layout) and pinned
//              in LDS the whole run; 6-member atomic barrier per (dir,btile).
//   K_out:     feats = concat(hf,hb) @ Wlin^T + blin (N padded 22->32)
// GEMM tile staging uses GLOBAL_LOAD_ASYNC_TO_LDS_B128 (ASYNCcnt path).
// ---------------------------------------------------------------------------

typedef _Float16 hf_t;
typedef __attribute__((ext_vector_type(16))) _Float16 v16h;
typedef __attribute__((ext_vector_type(8)))  float    v8f;
typedef unsigned int u32x4 __attribute__((ext_vector_type(4)));
typedef int          i32x8 __attribute__((ext_vector_type(8)));
typedef int          i32x4 __attribute__((ext_vector_type(4)));
typedef int          gv4i  __attribute__((vector_size(16)));   // builtin's pointee type

struct __align__(16) U4 { unsigned x, y, z, w; };
union Frag { v16h v; U4 q[2]; };

#define AS1 __attribute__((address_space(1)))
#define AS3 __attribute__((address_space(3)))

#if defined(__gfx1250__) && __has_builtin(__builtin_amdgcn_global_load_async_to_lds_b128)
#define HAVE_ASYNC_LDS 1
#else
#define HAVE_ASYNC_LDS 0
#endif
#if defined(__gfx1250__) && __has_builtin(__builtin_amdgcn_tensor_load_to_lds)
#define HAVE_TDM 1
#else
#define HAVE_TDM 0
#endif

// Generic->AS(3): low 32 bits of a generic LDS address are the LDS offset
// (ISA 10.2 aperture rules); generic->AS(1) is value-preserving.
// Builtin wants v4i32 pointees (signature vV4i*1 V4i*3 Ii Ii).
__device__ inline AS3 gv4i* lvec(void* p) { return (AS3 gv4i*)(unsigned)(size_t)p; }
__device__ inline AS1 gv4i* gvec(const void* p) { return (AS1 gv4i*)(size_t)p; }

#if defined(__gfx1250__) && __has_builtin(__builtin_amdgcn_s_wait_asynccnt)
#define ASYNC_WAIT() __builtin_amdgcn_s_wait_asynccnt(0)
#else
#define ASYNC_WAIT() asm volatile("s_wait_asynccnt 0" ::: "memory")
#endif
#if defined(__gfx1250__) && __has_builtin(__builtin_amdgcn_s_wait_tensorcnt)
#define TENSOR_WAIT() __builtin_amdgcn_s_wait_tensorcnt(0)
#else
#define TENSOR_WAIT() asm volatile("s_wait_tensorcnt 0" ::: "memory")
#endif

__device__ inline v8f wmma_f16(v16h a, v16h b, v8f c) {
  return __builtin_amdgcn_wmma_f32_16x16x32_f16(false, a, false, b, (short)0, c,
                                                false, false);
}

// A-matrix 16x32 f16 frag: lane L -> row L&15,
// 16 contiguous K at koff = (L>=16)?16:0.
__device__ inline v16h load_a(const hf_t* base, int ld, int lane, int kc) {
  const hf_t* p = base + (lane & 15) * ld + kc + ((lane >> 4) << 4);
  Frag f;
  f.q[0] = *(const U4*)(p);
  f.q[1] = *(const U4*)(p + 8);
  return f.v;
}
// B-matrix 32x16 f16 frag: lane L -> column L&15 (stored k-major),
// 16 contiguous K at koff = (L>=16)?16:0.
__device__ inline v16h load_b(const hf_t* base, int ld, int lane, int kc) {
  const hf_t* p = base + (lane & 15) * ld + kc + ((lane >> 4) << 4);
  Frag f;
  f.q[0] = *(const U4*)(p);
  f.q[1] = *(const U4*)(p + 8);
  return f.v;
}
// B frag from the TDM-padded Whh layout: row stride 432 halves; element k
// lives at k + 8*(k/64) (pad 4 DWORDs per 32 DWORDs stored). Each 16-half
// chunk stays inside one 64-half block -> 16B-aligned ds_load_b128.
__device__ inline v16h load_b_pad(const hf_t* rowbase, int lane, int kc) {
  int kk = kc + ((lane >> 4) << 4);
  int off = kk + ((kk >> 6) << 3);
  const hf_t* p = rowbase + (lane & 15) * 432 + off;
  Frag f;
  f.q[0] = *(const U4*)(p);
  f.q[1] = *(const U4*)(p + 8);
  return f.v;
}

// ------------------------------- prep kernels ------------------------------
__global__ void k_cvt(const float* __restrict__ src, hf_t* __restrict__ dst, int n) {
  int i = blockIdx.x * blockDim.x + threadIdx.x;
  if (i < n) dst[i] = (hf_t)src[i];
}
__global__ void k_wlin(const float* __restrict__ w, hf_t* __restrict__ dst) {
  int i = blockIdx.x * blockDim.x + threadIdx.x;   // 32*768
  if (i < 32 * 768) {
    int row = i / 768, col = i % 768;
    dst[i] = (row < 22) ? (hf_t)w[row * 768 + col] : (hf_t)0.0f;
  }
}
__global__ void k_bias(const float* __restrict__ bihf, const float* __restrict__ bhhf,
                       const float* __restrict__ bihb, const float* __restrict__ bhhb,
                       float* __restrict__ bias) {
  int i = blockIdx.x * blockDim.x + threadIdx.x;   // 2*1536
  if (i < 3072) {
    int d = i / 1536, j = i % 1536;
    bias[i] = d ? (bihb[j] + bhhb[j]) : (bihf[j] + bhhf[j]);
  }
}
__global__ void k_h0(const float* __restrict__ h0, hf_t* __restrict__ hbuf) {
  int i = blockIdx.x * blockDim.x + threadIdx.x;   // 2*64*384 (phase 0)
  if (i < 2 * 64 * 384) hbuf[i] = (hf_t)h0[i];
}
__global__ void k_meta(const int* __restrict__ start_ids, const int* __restrict__ masks,
                       int* __restrict__ meta) {
  int b = threadIdx.x;
  if (b >= 64) return;
  int n = 0, sent = 0;
  for (int s = 0; s < 512; ++s) {
    n    += (start_ids[b * 512 + s] >= 0);
    sent += (masks[b * 512 + s] != 0);
  }
  int last = start_ids[b * 512 + (n > 0 ? n - 1 : 0)];
  meta[b * 3 + 0] = n;
  meta[b * 3 + 1] = sent;
  meta[b * 3 + 2] = last;
}

// ------------------------------ align / gather -----------------------------
__global__ __launch_bounds__(256) void k_align(const float* __restrict__ hidden,
                                               const int* __restrict__ start_ids,
                                               const int* __restrict__ meta,
                                               hf_t* __restrict__ embeds) {
  const int t = blockIdx.x;   // 0..511
  const int b = blockIdx.y;   // 0..63
  const int n = meta[b * 3 + 0];
  const int sent = meta[b * 3 + 1];
  const int last = meta[b * 3 + 2];
  int idx;
  if (t == 0)      idx = 0;
  else if (t < n)  idx = start_ids[b * 512 + t] - 1;
  else if (t == n) idx = last;
  else             idx = 0;
  idx = idx < 0 ? 0 : (idx > 511 ? 511 : idx);
  const bool keep = t < sent;
  const float* src = hidden + ((size_t)b * 512 + idx) * 768;
  hf_t* dst = embeds + ((size_t)b * 512 + t) * 768;
  for (int d = threadIdx.x; d < 768; d += 256)
    dst[d] = keep ? (hf_t)src[d] : (hf_t)0.0f;
}

// --------------------- xg = embeds @ Wih^T + bias (WMMA) -------------------
__global__ __launch_bounds__(256) void k_gemm_xg(const hf_t* __restrict__ embeds,
                                                 const hf_t* __restrict__ wih,
                                                 const float* __restrict__ bias,
                                                 hf_t* __restrict__ xg) {
  __shared__ hf_t As[128][40];
  __shared__ hf_t Bs[64][40];
  const int tid = threadIdx.x, lane = tid & 31, w = tid >> 5;
  const int mt = blockIdx.x, nt = blockIdx.y, dir = blockIdx.z;
  const hf_t* wih_d = wih + (size_t)dir * 1536 * 768;
  const float* bias_d = bias + dir * 1536;
  hf_t* xg_d = xg + (size_t)dir * 32768 * 1536;
  const int wm = w & 3, wn = w >> 2;
  v8f acc[2][2] = {};
  const int ar = tid >> 1, ac = (tid & 1) * 16;
  const int br = tid >> 2, bc = (tid & 3) * 8;
  for (int kt = 0; kt < 24; ++kt) {
    const hf_t* ga = embeds + (size_t)(mt * 128 + ar) * 768 + kt * 32 + ac;
    const hf_t* gb = wih_d + (size_t)(nt * 64 + br) * 768 + kt * 32 + bc;
#if HAVE_ASYNC_LDS
    __builtin_amdgcn_global_load_async_to_lds_b128(gvec(ga),     lvec(&As[ar][ac]),     0, 0);
    __builtin_amdgcn_global_load_async_to_lds_b128(gvec(ga + 8), lvec(&As[ar][ac + 8]), 0, 0);
    __builtin_amdgcn_global_load_async_to_lds_b128(gvec(gb),     lvec(&Bs[br][bc]),     0, 0);
    ASYNC_WAIT();
#else
    *(U4*)&As[ar][ac]     = *(const U4*)(ga);
    *(U4*)&As[ar][ac + 8] = *(const U4*)(ga + 8);
    *(U4*)&Bs[br][bc]     = *(const U4*)(gb);
#endif
    if (kt + 1 < 24) { __builtin_prefetch(ga + 32, 0, 1); __builtin_prefetch(gb + 32, 0, 1); }
    __syncthreads();
    v16h a0 = load_a(&As[wm * 32][0],      40, lane, 0);
    v16h a1 = load_a(&As[wm * 32 + 16][0], 40, lane, 0);
    v16h b0 = load_b(&Bs[wn * 32][0],      40, lane, 0);
    v16h b1 = load_b(&Bs[wn * 32 + 16][0], 40, lane, 0);
    acc[0][0] = wmma_f16(a0, b0, acc[0][0]);
    acc[0][1] = wmma_f16(a0, b1, acc[0][1]);
    acc[1][0] = wmma_f16(a1, b0, acc[1][0]);
    acc[1][1] = wmma_f16(a1, b1, acc[1][1]);
    __syncthreads();
  }
  for (int i2 = 0; i2 < 2; ++i2)
    for (int j2 = 0; j2 < 2; ++j2)
      for (int r = 0; r < 8; ++r) {
        int m = mt * 128 + wm * 32 + i2 * 16 + r + ((lane >> 4) << 3);
        int n = nt * 64 + wn * 32 + j2 * 16 + (lane & 15);
        xg_d[(size_t)m * 1536 + n] = (hf_t)(acc[i2][j2][r] + bias_d[n]);
      }
}

// ------------------------------ LSTM recurrence ----------------------------
// 48 WGs: dir(2) x batch-tile(4, M=16) x h-chunk(6, 64 h-elems). Each WG owns
// gate cols {g*384 + hc*64 .. +64 | g=0..3}; Whh slice (256 rows x 384 k f16,
// padded to 432/row = 216KB) pinned in LDS, loaded via TDM.
__global__ __launch_bounds__(256) void k_scan(const hf_t* __restrict__ whh,
                                              const hf_t* __restrict__ xg,
                                              hf_t* __restrict__ hbuf,
                                              hf_t* __restrict__ hcat,
                                              const float* __restrict__ c0,
                                              unsigned* __restrict__ bar) {
  extern __shared__ char smem[];
  hf_t* Whs = (hf_t*)smem;                                   // 256 rows x 432
  float (*gbuf)[260] = (float(*)[260])(smem + 256 * 432 * 2);
  float (*cbuf)[64]  = (float(*)[64])(smem + 256 * 432 * 2 + 16 * 260 * 4);
  const int tid = threadIdx.x, lane = tid & 31, w = tid >> 5;
  int x = blockIdx.x;
  const int dir = x / 24; x %= 24;
  const int btile = x / 6;
  const int hc = x % 6;
  const int grp = dir * 4 + btile;
  const hf_t* whh_d = whh + (size_t)dir * 1536 * 384;

#if HAVE_TDM
  if (tid < 32) {  // one wave issues 4 TDM descriptors (one per gate block)
    for (int g = 0; g < 4; ++g) {
      size_t gaddr = (size_t)(whh_d + (size_t)(g * 384 + hc * 64) * 384);
      unsigned ldsa = (unsigned)(size_t)&Whs[g * 64 * 432];
      u32x4 g0 = {1u,                                   // count=1, user D#
                  ldsa,                                 // lds_addr
                  (unsigned)(gaddr & 0xFFFFFFFFu),
                  (unsigned)((gaddr >> 32) & 0x01FFFFFFu) | (2u << 30)}; // type=2
      // data_size=2B, pad_enable, pad_interval=32 DW, pad_amount=4 DW,
      // tensor_dim0=384, tensor_dim1=64, tile_dim0=384, tile_dim1=64,
      // tensor_dim0_stride=384.
      i32x8 g1 = {(int)0x07110000, (int)(384u << 16), (int)(64u << 16),
                  (int)(384u << 16), 64, 384, 0, 0};
      i32x4 gz = {0, 0, 0, 0};
#if __clang_major__ >= 23
      i32x8 gz8 = {0, 0, 0, 0, 0, 0, 0, 0};
      __builtin_amdgcn_tensor_load_to_lds(g0, g1, gz, gz, gz8, 0);
#else
      __builtin_amdgcn_tensor_load_to_lds(g0, g1, gz, gz, 0);
#endif
    }
    TENSOR_WAIT();
  }
#else
  {  // fallback: manual copy into the same padded layout
    const int g = tid >> 6, j = tid & 63;
    const hf_t* src = whh_d + (size_t)(g * 384 + hc * 64 + j) * 384;
    hf_t* drow = Whs + (size_t)tid * 432;
    for (int k = 0; k < 384; k += 8)
      *(U4*)&drow[k + ((k >> 6) << 3)] = *(const U4*)(src + k);
  }
#endif
  for (int q = 0; q < 4; ++q) {  // init cell state slice
    int idx = tid * 4 + q;
    int row = idx >> 6, j = idx & 63;
    cbuf[row][j] = c0[(size_t)(dir * 64 + btile * 16 + row) * 384 + hc * 64 + j];
  }
  __syncthreads();

  const hf_t* xg_d = xg + (size_t)dir * 32768 * 1536;
  unsigned* mybar = bar + grp * 16;
  for (int t = 0; t < 512; ++t) {
    const int tx = dir ? (511 - t) : t;
    const hf_t* hA = hbuf + (size_t)(((t & 1) * 2 + dir) * 64 + btile * 16) * 384;
    for (int ni = 0; ni < 2; ++ni) {      // wave covers 2 of 16 n-tiles (N=256)
      const int ntile = w * 2 + ni;
      v8f acc = {};
      for (int kc = 0; kc < 384; kc += 32) {
        v16h af = load_a(hA, 384, lane, kc);
        v16h bf = load_b_pad(Whs + (size_t)(ntile * 16) * 432, lane, kc);
        acc = wmma_f16(af, bf, acc);
      }
      for (int r = 0; r < 8; ++r) {
        int row  = r + ((lane >> 4) << 3);
        int col  = ntile * 16 + (lane & 15);
        int gcol = (col >> 6) * 384 + hc * 64 + (col & 63);
        int bb   = btile * 16 + row;
        float xv = (float)xg_d[(size_t)(bb * 512 + tx) * 1536 + gcol];
        gbuf[row][col] = acc[r] + xv;
      }
    }
    __syncthreads();
    hf_t* hNext = hbuf + (size_t)((((t + 1) & 1) * 2 + dir) * 64) * 384;
    for (int q = 0; q < 4; ++q) {
      int idx = tid * 4 + q;
      int row = idx >> 6, j = idx & 63;
      float gi = gbuf[row][j];
      float gf = gbuf[row][64 + j];
      float gg = gbuf[row][128 + j];
      float go = gbuf[row][192 + j];
      float c  = cbuf[row][j];
      float si = 1.0f / (1.0f + __expf(-gi));
      float sf = 1.0f / (1.0f + __expf(-gf));
      float so = 1.0f / (1.0f + __expf(-go));
      float nc = sf * c + si * tanhf(gg);
      float nh = so * tanhf(nc);
      cbuf[row][j] = nc;
      int bb = btile * 16 + row;
      hNext[(size_t)bb * 384 + hc * 64 + j] = (hf_t)nh;
      hcat[(size_t)(bb * 512 + tx) * 768 + dir * 384 + hc * 64 + j] = (hf_t)nh;
    }
    __threadfence();       // make this WG's h visible device-wide
    __syncthreads();
    if (tid == 0) {        // 6-member group barrier, monotonic counter
      atomicAdd(mybar, 1u);
      unsigned target = (unsigned)(t + 1) * 6u;
      while (atomicAdd(mybar, 0u) < target) { __builtin_amdgcn_s_sleep(1); }
      __threadfence();
    }
    __syncthreads();
  }
}

// ------------------ feats = concat(hf,hb) @ Wlin^T + blin ------------------
__global__ __launch_bounds__(256) void k_out(const hf_t* __restrict__ hcat,
                                             const hf_t* __restrict__ wlin,
                                             const float* __restrict__ blin,
                                             float* __restrict__ out) {
  __shared__ hf_t As[128][40];
  __shared__ hf_t Bs[32][40];
  const int tid = threadIdx.x, lane = tid & 31, w = tid >> 5;
  const int mt = blockIdx.x;
  v8f acc[2] = {};
  const int ar = tid >> 1, ac = (tid & 1) * 16;
  for (int kt = 0; kt < 24; ++kt) {
    const hf_t* ga = hcat + (size_t)(mt * 128 + ar) * 768 + kt * 32 + ac;
#if HAVE_ASYNC_LDS
    __builtin_amdgcn_global_load_async_to_lds_b128(gvec(ga),     lvec(&As[ar][ac]),     0, 0);
    __builtin_amdgcn_global_load_async_to_lds_b128(gvec(ga + 8), lvec(&As[ar][ac + 8]), 0, 0);
    if (tid < 128) {
      int br = tid >> 2, bc = (tid & 3) * 8;
      __builtin_amdgcn_global_load_async_to_lds_b128(
          gvec(wlin + (size_t)br * 768 + kt * 32 + bc), lvec(&Bs[br][bc]), 0, 0);
    }
    ASYNC_WAIT();
#else
    *(U4*)&As[ar][ac]     = *(const U4*)(ga);
    *(U4*)&As[ar][ac + 8] = *(const U4*)(ga + 8);
    if (tid < 128) {
      int br = tid >> 2, bc = (tid & 3) * 8;
      *(U4*)&Bs[br][bc] = *(const U4*)(wlin + (size_t)br * 768 + kt * 32 + bc);
    }
#endif
    if (kt + 1 < 24) __builtin_prefetch(ga + 32, 0, 1);
    __syncthreads();
    v16h a  = load_a(&As[w * 16][0], 40, lane, 0);
    v16h b0 = load_b(&Bs[0][0],  40, lane, 0);
    v16h b1 = load_b(&Bs[16][0], 40, lane, 0);
    acc[0] = wmma_f16(a, b0, acc[0]);
    acc[1] = wmma_f16(a, b1, acc[1]);
    __syncthreads();
  }
  for (int j2 = 0; j2 < 2; ++j2)
    for (int r = 0; r < 8; ++r) {
      int n = j2 * 16 + (lane & 15);
      if (n < 22) {
        int m = mt * 128 + w * 16 + r + ((lane >> 4) << 3);
        out[(size_t)m * 22 + n] = acc[j2][r] + blin[n];
      }
    }
}

// --------------------------------- launcher --------------------------------
extern "C" void kernel_launch(void* const* d_in, const int* in_sizes, int n_in,
                              void* d_out, int out_size, void* d_ws, size_t ws_size,
                              hipStream_t stream) {
  const float* hidden = (const float*)d_in[0];
  const float* h0     = (const float*)d_in[1];
  const float* c0     = (const float*)d_in[2];
  const float* Wihf   = (const float*)d_in[3];
  const float* Whhf   = (const float*)d_in[4];
  const float* bihf   = (const float*)d_in[5];
  const float* bhhf   = (const float*)d_in[6];
  const float* Wihb   = (const float*)d_in[7];
  const float* Whhb   = (const float*)d_in[8];
  const float* bihb   = (const float*)d_in[9];
  const float* bhhb   = (const float*)d_in[10];
  const float* Wlin   = (const float*)d_in[11];
  const float* blin   = (const float*)d_in[12];
  const int* start_ids = (const int*)d_in[13];
  const int* masks     = (const int*)d_in[14];
  float* out = (float*)d_out;

  size_t off = 0;
  char* base = (char*)d_ws;
  auto carve = [&](size_t bytes) -> void* {
    void* r = base + off;
    off = (off + bytes + 255) & ~(size_t)255;
    return r;
  };
  hf_t* embeds  = (hf_t*)carve((size_t)64 * 512 * 768 * 2);      // 50.3 MB
  hf_t* wih_hf  = (hf_t*)carve((size_t)2 * 1536 * 768 * 2);      // 4.7 MB
  hf_t* whh_hf  = (hf_t*)carve((size_t)2 * 1536 * 384 * 2);      // 2.4 MB
  hf_t* wlin_hf = (hf_t*)carve((size_t)32 * 768 * 2);            // 48 KB
  float* bias   = (float*)carve((size_t)3072 * 4);
  hf_t* xg      = (hf_t*)carve((size_t)2 * 32768 * 1536 * 2);    // 201 MB
  hf_t* hcat    = (hf_t*)carve((size_t)32768 * 768 * 2);         // 50.3 MB
  hf_t* hbuf    = (hf_t*)carve((size_t)2 * 2 * 64 * 384 * 2);    // ping-pong h
  unsigned* bar = (unsigned*)carve((size_t)8 * 16 * 4);
  int* meta     = (int*)carve((size_t)64 * 3 * 4);

  (void)hipMemsetAsync(bar, 0, 8 * 16 * 4, stream);

  k_cvt<<<(1536 * 768 + 255) / 256, 256, 0, stream>>>(Wihf, wih_hf, 1536 * 768);
  k_cvt<<<(1536 * 768 + 255) / 256, 256, 0, stream>>>(Wihb, wih_hf + (size_t)1536 * 768, 1536 * 768);
  k_cvt<<<(1536 * 384 + 255) / 256, 256, 0, stream>>>(Whhf, whh_hf, 1536 * 384);
  k_cvt<<<(1536 * 384 + 255) / 256, 256, 0, stream>>>(Whhb, whh_hf + (size_t)1536 * 384, 1536 * 384);
  k_wlin<<<(32 * 768 + 255) / 256, 256, 0, stream>>>(Wlin, wlin_hf);
  k_bias<<<(3072 + 255) / 256, 256, 0, stream>>>(bihf, bhhf, bihb, bhhb, bias);
  k_h0<<<(2 * 64 * 384 + 255) / 256, 256, 0, stream>>>(h0, hbuf);
  k_meta<<<1, 64, 0, stream>>>(start_ids, masks, meta);
  k_align<<<dim3(512, 64), 256, 0, stream>>>(hidden, start_ids, meta, embeds);
  k_gemm_xg<<<dim3(256, 24, 2), 256, 0, stream>>>(embeds, wih_hf, bias, xg);

  const size_t scan_lds = 256 * 432 * 2 + 16 * 260 * 4 + 16 * 64 * 4;  // 241,920 B
  k_scan<<<48, 256, scan_lds, stream>>>(whh_hf, xg, hbuf, hcat, c0, bar);

  k_out<<<256, 256, 0, stream>>>(hcat, wlin_hf, blin, out);
}
